// MultiHeadAttentionBlock_2052994367505
// MI455X (gfx1250) — compile-verified, matrix operand order checked by reading, not run on hardware
//
#include <hip/hip_runtime.h>
#include <hip/hip_bf16.h>

// ---------------------------------------------------------------------------
// Problem constants (from reference): B=32, C=128, D_IN=1024, HEADS=128,
// K_SIZE=64, V_SIZE=8, OUT=1024.  Rows = B*C = 4096.
// ---------------------------------------------------------------------------

typedef __bf16 bf16_t;
typedef __attribute__((ext_vector_type(16))) __bf16 v16bf;
typedef __attribute__((ext_vector_type(8)))  __bf16 v8bf;
typedef __attribute__((ext_vector_type(8)))  float  v8f;

union Frag {
    v16bf v;
    v8bf  h[2];
};

// ---------------------------------------------------------------------------
// CDNA5 async global->LDS copy (ASYNCcnt-tracked) + wait.  Inline asm is
// portable across ROCm-7.2 / amdgpu-toolchain (builtin arity differs).
// LDS byte offset = low 32 bits of the flat shared-memory address.
// ---------------------------------------------------------------------------
__device__ __forceinline__ void async_ld16(unsigned lds_off, const void* g) {
    asm volatile("global_load_async_to_lds_b128 %0, %1, off"
                 :: "v"(lds_off), "v"(g) : "memory");
}
__device__ __forceinline__ void wait_async0() {
    asm volatile("s_wait_asynccnt 0x0" ::: "memory");
}

// ---------------------------------------------------------------------------
// f32 -> bf16 elementwise conversion
// ---------------------------------------------------------------------------
__global__ void cvt_f32_to_bf16(const float* __restrict__ src,
                                bf16_t* __restrict__ dst, int n) {
    int i = blockIdx.x * blockDim.x + threadIdx.x;
    if (i < n) dst[i] = (bf16_t)src[i];
}

// ---------------------------------------------------------------------------
// WMMA GEMM:  C[M,N] = A[M,Kd] * B[N,Kd]^T + bias[N]
//
// Block tile 128(M) x 128(N), K staged in chunks of 32 through LDS using
// double-buffered async global->LDS copies.  8 waves arranged 4(M) x 2(N);
// each wave computes a 32x64 tile = 2x4 WMMA accumulators (A frag reused 4x,
// B frag reused 2x).  LDS rows padded to 80B for conflict-free ds_load_b128.
// grid = (N/128, M/128);  outBf16: 1 -> store bf16, 0 -> store f32.
// ---------------------------------------------------------------------------
constexpr int KB    = 32;          // K elements per stage (one WMMA k-step)
constexpr int ROWB  = 80;          // padded LDS row stride in bytes (64B data)
constexpr int TILEB = 128 * ROWB;  // 10240 B per staged tile

__global__ void gemm_bf16_wmma(const bf16_t* __restrict__ A,
                               const bf16_t* __restrict__ B,
                               const float*  __restrict__ bias,
                               void* __restrict__ Cout,
                               int M, int N, int Kd, int outBf16) {
    __shared__ __align__(16) unsigned char sMem[2][2][TILEB];  // [buf][A|B]

    const int tid  = threadIdx.x;
    const int wave = tid >> 5;
    const int lane = tid & 31;
    const int wm   = wave >> 1;          // 0..3  (M direction)
    const int wn   = wave & 1;           // 0..1  (N direction)
    const int m0   = blockIdx.y * 128;
    const int n0   = blockIdx.x * 128;

    // WMMA 16-bit A/B per-lane layout: lane L holds row/col (L&15) with
    // K-chunks [0,8)+[16,24) (L<16) or [8,16)+[24,32) (L>=16).
    const int mrow  = lane & 15;
    const int koffb = ((lane >> 4) << 3) * 2;   // 0 or 16 (bytes)
    const int rbase = (lane >> 4) << 3;

    // async-fill assignment: 512 16-byte chunks per tile, 2 per thread.
    const int fr0 = tid >> 2;            // rows 0..63
    const int fcb = (tid & 3) * 16;      // byte offset within 64B row chunk

    auto fill = [&](int buf, int ks) {
        const bf16_t* Ag = A + (size_t)m0 * Kd + ks * KB;
        const bf16_t* Bg = B + (size_t)n0 * Kd + ks * KB;
        unsigned sa = (unsigned)(uintptr_t)&sMem[buf][0][0];
        unsigned sb = (unsigned)(uintptr_t)&sMem[buf][1][0];
        async_ld16(sa + fr0 * ROWB + fcb,
                   (const char*)(Ag + (size_t)fr0 * Kd) + fcb);
        async_ld16(sa + (fr0 + 64) * ROWB + fcb,
                   (const char*)(Ag + (size_t)(fr0 + 64) * Kd) + fcb);
        async_ld16(sb + fr0 * ROWB + fcb,
                   (const char*)(Bg + (size_t)fr0 * Kd) + fcb);
        async_ld16(sb + (fr0 + 64) * ROWB + fcb,
                   (const char*)(Bg + (size_t)(fr0 + 64) * Kd) + fcb);
    };

    v8f acc[2][4];
    #pragma unroll
    for (int s = 0; s < 2; ++s)
        #pragma unroll
        for (int t = 0; t < 4; ++t) acc[s][t] = (v8f){};

    const int NK = Kd / KB;
    int buf = 0;
    fill(0, 0);
    wait_async0();
    __syncthreads();

    for (int ks = 0; ks < NK; ++ks) {
        if (ks + 1 < NK) fill(buf ^ 1, ks + 1);   // prefetch next stage

        const unsigned char* sa = &sMem[buf][0][0];
        const unsigned char* sb = &sMem[buf][1][0];

        Frag af[2];
        #pragma unroll
        for (int s = 0; s < 2; ++s) {
            const unsigned char* rp = sa + (wm * 32 + s * 16 + mrow) * ROWB;
            af[s].h[0] = *(const v8bf*)(rp + koffb);
            af[s].h[1] = *(const v8bf*)(rp + 32 + koffb);
        }
        #pragma unroll
        for (int t = 0; t < 4; ++t) {
            const unsigned char* rp = sb + (wn * 64 + t * 16 + mrow) * ROWB;
            Frag bf;
            bf.h[0] = *(const v8bf*)(rp + koffb);
            bf.h[1] = *(const v8bf*)(rp + 32 + koffb);
            #pragma unroll
            for (int s = 0; s < 2; ++s)
                acc[s][t] = __builtin_amdgcn_wmma_f32_16x16x32_bf16(
                    false, af[s].v, false, bf.v, (short)0, acc[s][t],
                    false, false);
        }

        wait_async0();      // our prefetch chunks landed
        __syncthreads();    // everyone done reading buf / writing buf^1
        buf ^= 1;
    }

    // C/D layout: lane L holds column n=(L&15); VGPR r holds row
    // m = r + (L>=16 ? 8 : 0).
    #pragma unroll
    for (int s = 0; s < 2; ++s) {
        const int mb = m0 + wm * 32 + s * 16 + rbase;
        #pragma unroll
        for (int t = 0; t < 4; ++t) {
            const int ncol = n0 + wn * 64 + t * 16 + (lane & 15);
            const float bval = bias[ncol];
            if (outBf16) {
                bf16_t* C = (bf16_t*)Cout;
                #pragma unroll
                for (int r = 0; r < 8; ++r)
                    C[(size_t)(mb + r) * N + ncol] = (bf16_t)(acc[s][t][r] + bval);
            } else {
                float* C = (float*)Cout;
                #pragma unroll
                for (int r = 0; r < 8; ++r)
                    C[(size_t)(mb + r) * N + ncol] = acc[s][t][r] + bval;
            }
        }
    }
}

// ---------------------------------------------------------------------------
// Fused attention: one workgroup per (b,h) flat index bh in [0,4096).
//   Qrow = Qb[bh] viewed [128][64], Krow likewise.
//   S = Q K^T / 8         (WMMA bf16, result into 64KB LDS)
//   att = softmax over the QUERY axis (column softmax of S)
//   ctx[f] = sum_j att[f>>3][j] * Vrow[j*8 + (f&7)],  f in [0,1024)
// ---------------------------------------------------------------------------
__global__ void attn_fused(const bf16_t* __restrict__ Qb,
                           const bf16_t* __restrict__ Kb,
                           const float*  __restrict__ Vf,
                           bf16_t* __restrict__ Ctx) {
    __shared__ float S[128 * 128];   // 64 KB

    const int bh = blockIdx.x;
    const bf16_t* Qrow = Qb + (size_t)bh * 8192;
    const bf16_t* Krow = Kb + (size_t)bh * 8192;
    const float*  Vrow = Vf + (size_t)bh * 1024;
    bf16_t*       Crow = Ctx + (size_t)bh * 1024;

    const int tid  = threadIdx.x;
    const int wave = tid >> 5;
    const int lane = tid & 31;
    const int mrow = lane & 15;
    const int koff = (lane >> 4) << 3;
    const int rbase = (lane >> 4) << 3;

    // --- scores: wave w computes S rows [16w, 16w+16), all 8 column tiles
    const int m0 = wave * 16;
    for (int nt = 0; nt < 8; ++nt) {
        v8f acc = (v8f){};
        #pragma unroll
        for (int k0 = 0; k0 < 64; k0 += 32) {
            Frag a, b;
            a.h[0] = *(const v8bf*)(Qrow + (m0 + mrow) * 64 + k0 + koff);
            a.h[1] = *(const v8bf*)(Qrow + (m0 + mrow) * 64 + k0 + 16 + koff);
            b.h[0] = *(const v8bf*)(Krow + (nt * 16 + mrow) * 64 + k0 + koff);
            b.h[1] = *(const v8bf*)(Krow + (nt * 16 + mrow) * 64 + k0 + 16 + koff);
            acc = __builtin_amdgcn_wmma_f32_16x16x32_bf16(
                false, a.v, false, b.v, (short)0, acc, false, false);
        }
        const int ncol = nt * 16 + (lane & 15);
        #pragma unroll
        for (int r = 0; r < 8; ++r)
            S[(m0 + rbase + r) * 128 + ncol] = acc[r] * 0.125f;
    }
    __syncthreads();

    // --- softmax over the query axis: thread j normalizes column j
    if (tid < 128) {
        const int j = tid;
        float mx = -3.402823466e38f;
        for (int i = 0; i < 128; ++i) mx = fmaxf(mx, S[i * 128 + j]);
        float sum = 0.f;
        for (int i = 0; i < 128; ++i) {
            float e = __expf(S[i * 128 + j] - mx);
            S[i * 128 + j] = e;
            sum += e;
        }
        const float inv = 1.0f / sum;
        for (int i = 0; i < 128; ++i) S[i * 128 + j] *= inv;
    }
    __syncthreads();

    // --- ctx: 1024 outputs, 4 per thread
    #pragma unroll
    for (int q = 0; q < 4; ++q) {
        const int f  = tid + 256 * q;
        const int m  = f >> 3;
        const int vv = f & 7;
        float a = 0.f;
        for (int j = 0; j < 128; ++j)
            a += S[m * 128 + j] * Vrow[j * 8 + vv];
        Crow[f] = (bf16_t)a;
    }
}

// ---------------------------------------------------------------------------
// Workspace layout (bytes)
// ---------------------------------------------------------------------------
constexpr size_t OFF_XB  = 0;                        // x bf16      [4096,1024]  8 MB
constexpr size_t OFF_WQB = OFF_XB  + 8388608;        // wq bf16     [8192,1024] 16 MB
constexpr size_t OFF_WKB = OFF_WQB + 16777216;       // wk bf16     [8192,1024] 16 MB
constexpr size_t OFF_WVB = OFF_WKB + 16777216;       // wv bf16     [1024,1024]  2 MB
constexpr size_t OFF_WOB = OFF_WVB + 2097152;        // wo bf16     [1024,1024]  2 MB
constexpr size_t OFF_QB  = OFF_WOB + 2097152;        // qlin bf16   [4096,8192] 64 MB
constexpr size_t OFF_KB  = OFF_QB  + 67108864;       // klin bf16   [4096,8192] 64 MB
constexpr size_t OFF_VF  = OFF_KB  + 67108864;       // vlin f32    [4096,1024] 16 MB
constexpr size_t OFF_CTX = OFF_VF  + 16777216;       // ctx bf16    [4096,1024]  8 MB
// total = OFF_CTX + 8388608 = 205,520,896 bytes

extern "C" void kernel_launch(void* const* d_in, const int* in_sizes, int n_in,
                              void* d_out, int out_size, void* d_ws, size_t ws_size,
                              hipStream_t stream) {
    const float* x  = (const float*)d_in[0];
    const float* wq = (const float*)d_in[1];
    const float* bq = (const float*)d_in[2];
    const float* wk = (const float*)d_in[3];
    const float* bk = (const float*)d_in[4];
    const float* wv = (const float*)d_in[5];
    const float* bv = (const float*)d_in[6];
    const float* wo = (const float*)d_in[7];
    const float* bo = (const float*)d_in[8];
    float* out = (float*)d_out;

    char* ws = (char*)d_ws;
    bf16_t* xb  = (bf16_t*)(ws + OFF_XB);
    bf16_t* wqb = (bf16_t*)(ws + OFF_WQB);
    bf16_t* wkb = (bf16_t*)(ws + OFF_WKB);
    bf16_t* wvb = (bf16_t*)(ws + OFF_WVB);
    bf16_t* wob = (bf16_t*)(ws + OFF_WOB);
    bf16_t* Qb  = (bf16_t*)(ws + OFF_QB);
    bf16_t* Kb  = (bf16_t*)(ws + OFF_KB);
    float*  Vf  = (float*) (ws + OFF_VF);
    bf16_t* Ctx = (bf16_t*)(ws + OFF_CTX);

    // --- 1) convert inputs to bf16
    cvt_f32_to_bf16<<<(4194304 + 255) / 256, 256, 0, stream>>>(x,  xb,  4194304);
    cvt_f32_to_bf16<<<(8388608 + 255) / 256, 256, 0, stream>>>(wq, wqb, 8388608);
    cvt_f32_to_bf16<<<(8388608 + 255) / 256, 256, 0, stream>>>(wk, wkb, 8388608);
    cvt_f32_to_bf16<<<(1048576 + 255) / 256, 256, 0, stream>>>(wv, wvb, 1048576);
    cvt_f32_to_bf16<<<(1048576 + 255) / 256, 256, 0, stream>>>(wo, wob, 1048576);

    // --- 2) projections: qlin/klin [4096,8192] bf16, vlin [4096,1024] f32
    gemm_bf16_wmma<<<dim3(8192 / 128, 4096 / 128), 256, 0, stream>>>(
        xb, wqb, bq, Qb, 4096, 8192, 1024, /*outBf16=*/1);
    gemm_bf16_wmma<<<dim3(8192 / 128, 4096 / 128), 256, 0, stream>>>(
        xb, wkb, bk, Kb, 4096, 8192, 1024, /*outBf16=*/1);
    gemm_bf16_wmma<<<dim3(1024 / 128, 4096 / 128), 256, 0, stream>>>(
        xb, wvb, bv, Vf, 4096, 1024, 1024, /*outBf16=*/0);

    // --- 3) fused scores + query-axis softmax + ctx, one block per (b,h)
    attn_fused<<<4096, 256, 0, stream>>>(Qb, Kb, Vf, Ctx);

    // --- 4) output projection: out = ctx @ wo^T + bo  (f32)
    gemm_bf16_wmma<<<dim3(1024 / 128, 4096 / 128), 256, 0, stream>>>(
        Ctx, wob, bo, out, 4096, 1024, 1024, /*outBf16=*/0);
}
